// EdgePointGNN_83528523973396
// MI455X (gfx1250) — compile-verified, matrix operand order checked by reading, not run on hardware
//
#include <hip/hip_runtime.h>

typedef __attribute__((ext_vector_type(16))) _Float16 v16h;
typedef __attribute__((ext_vector_type(8)))  _Float16 v8h;
typedef __attribute__((ext_vector_type(8)))  float    v8f;
typedef __attribute__((ext_vector_type(4)))  float    v4f;

#define WAVES 8
#define LN_EPS 1e-5f

// ---------------- WMMA fragment loaders ----------------
// A: 16x32 (MxK) from row-major [16][128] f16 tile in LDS.
// lane l (r=l&15, h=l>>4): elems 0..7  = K koff+h*8 .. +7   (contiguous)
//                          elems 8..15 = K koff+16+h*8 .. +7 (contiguous)
__device__ inline v16h frag_A(const _Float16* __restrict__ s, int koff, int lane) {
  int r = lane & 15, h = lane >> 4;
  const _Float16* base = s + r * 128 + koff + h * 8;
  v8h lo = *(const v8h*)(base);
  v8h hi = *(const v8h*)(base + 16);
  v16h a;
#pragma unroll
  for (int i = 0; i < 8; ++i) { a[i] = lo[i]; a[i + 8] = hi[i]; }
  return a;
}

// B: pre-swizzled weights: [tile][lane][16 halfs] contiguous -> two b128 LDS loads.
__device__ inline v16h frag_B_sw(const _Float16* __restrict__ w, int tile, int lane) {
  const _Float16* base = w + ((tile * 32 + lane) << 4);
  v8h lo = *(const v8h*)(base);
  v8h hi = *(const v8h*)(base + 8);
  v16h b;
#pragma unroll
  for (int i = 0; i < 8; ++i) { b[i] = lo[i]; b[i + 8] = hi[i]; }
  return b;
}

// LayerNorm + ReLU over one 128-wide row; f32 in (LDS), f16 out (LDS, row-major).
// Single-pass moment stats; fully vectorized loads/stores.
__device__ inline void ln_relu_row_128(const float* __restrict__ hr, _Float16* __restrict__ row,
                                       const float* __restrict__ g, const float* __restrict__ be) {
  float s1 = 0.f, s2 = 0.f;
#pragma unroll
  for (int c = 0; c < 128; c += 4) {
    v4f t = *(const v4f*)(hr + c);
#pragma unroll
    for (int j = 0; j < 4; ++j) { s1 += t[j]; s2 += t[j] * t[j]; }
  }
  float m = s1 * (1.0f / 128.0f);
  float var = s2 * (1.0f / 128.0f) - m * m;
  float inv = rsqrtf(fmaxf(var, 0.f) + LN_EPS);
#pragma unroll
  for (int c = 0; c < 128; c += 8) {
    v4f t0 = *(const v4f*)(hr + c);
    v4f t1 = *(const v4f*)(hr + c + 4);
    v4f G0 = *(const v4f*)(g + c);
    v4f G1 = *(const v4f*)(g + c + 4);
    v4f B0 = *(const v4f*)(be + c);
    v4f B1 = *(const v4f*)(be + c + 4);
    v8h o;
#pragma unroll
    for (int j = 0; j < 4; ++j) {
      o[j]     = (_Float16)fmaxf((t0[j] - m) * inv * G0[j] + B0[j], 0.0f);
      o[4 + j] = (_Float16)fmaxf((t1[j] - m) * inv * G1[j] + B1[j], 0.0f);
    }
    *(v8h*)(row + c) = o;
  }
}

// ---------------- Edge-MLP layer ----------------
// CIN: input channels; KT1: K-tiles (of 32) covering the layer-1 input.
// Wave processes 32 edges (two 16-row M-tiles) per iteration.
// For CIN==64 the input layout is PERMUTED (weights permuted to match):
//   cols 0..63 = xi[0..63], col 64 = dist2, cols 65..127 = xj[1..63]
//   (xj[1],xj[2] get zero weights; reference cols remapped in cvt_sw_f16 mode 1)
template <int CIN, int KT1>
__global__ __launch_bounds__(WAVES * 32, 1)
void edge_layer(const float* __restrict__ x,
                const int* __restrict__ src, const int* __restrict__ dst,
                const _Float16* __restrict__ W1s, const float* __restrict__ b1,
                const float* __restrict__ g1,   const float* __restrict__ be1,
                const _Float16* __restrict__ W2s, const float* __restrict__ b2,
                const float* __restrict__ g2,   const float* __restrict__ be2,
                const _Float16* __restrict__ W3s, const float* __restrict__ b3,
                float* __restrict__ out, int E) {
  constexpr int DIN = 2 * CIN - 2;
  constexpr int W1H = KT1 * 8 * 512;
  constexpr int W2H = 4 * 8 * 512;
  constexpr int W3H = 4 * 4 * 512;
  extern __shared__ char smem[];
  _Float16* sW1 = (_Float16*)smem;
  _Float16* sW2 = sW1 + W1H;
  _Float16* sW3 = sW2 + W2H;
  float* fb   = (float*)(sW3 + W3H);
  float* sb1  = fb;         float* sg1 = sb1 + 128;  float* sbe1 = sg1 + 128;
  float* sb2  = sbe1 + 128; float* sg2 = sb2 + 128;  float* sbe2 = sg2 + 128;
  float* sb3  = sbe2 + 128;                          // [64]
  float* shrow_all = sb3 + 64;                       // [WAVES][32*128] f32
  _Float16* sa_all = (_Float16*)(shrow_all + WAVES * 32 * 128); // [WAVES][32*128] f16
  int* sdst_all = (int*)(sa_all + WAVES * 32 * 128);            // [WAVES][32]

  int tid = threadIdx.x;
  {
    const uint4* gw1 = (const uint4*)W1s; uint4* lw1 = (uint4*)sW1;
    for (int i = tid; i < W1H / 8; i += blockDim.x) lw1[i] = gw1[i];
    const uint4* gw2 = (const uint4*)W2s; uint4* lw2 = (uint4*)sW2;
    for (int i = tid; i < W2H / 8; i += blockDim.x) lw2[i] = gw2[i];
    const uint4* gw3 = (const uint4*)W3s; uint4* lw3 = (uint4*)sW3;
    for (int i = tid; i < W3H / 8; i += blockDim.x) lw3[i] = gw3[i];
  }
  for (int i = tid; i < 128; i += blockDim.x) {
    sb1[i] = b1[i]; sg1[i] = g1[i]; sbe1[i] = be1[i];
    sb2[i] = b2[i]; sg2[i] = g2[i]; sbe2[i] = be2[i];
  }
  for (int i = tid; i < 64; i += blockDim.x) sb3[i] = b3[i];
  __syncthreads();

  int wave = tid >> 5, lane = tid & 31;
  float*    shrow = shrow_all + wave * 32 * 128;
  _Float16* sa    = sa_all    + wave * 32 * 128;
  int*      sdst  = sdst_all  + wave * 32;
  int nc = lane & 15, hh = lane >> 4;

  int ngroups = (E + 31) >> 5;
  for (int t = blockIdx.x * WAVES + wave; t < ngroups; t += gridDim.x * WAVES) {
    int e0 = t << 5;
    // ---- stage 32 gathered edges as f16 rows (one edge per lane) ----
    {
      int e = e0 + lane;
      _Float16* row = sa + lane * 128;
      if (e < E) {
        int si = src[e], di = dst[e];
        sdst[lane] = di;
        const float* xi = x + (long)di * CIN;
        const float* xj = x + (long)si * CIN;
        float d2 = 0.f;
#pragma unroll
        for (int k = 0; k < 3; ++k) { float d = xj[k] - xi[k]; d2 += d * d; }
        if (CIN == 64) {
#pragma unroll
          for (int c = 0; c < 64; c += 8) {
            v4f a0 = *(const v4f*)(xi + c);
            v4f a1 = *(const v4f*)(xi + c + 4);
            v8h o;
#pragma unroll
            for (int j = 0; j < 4; ++j) { o[j] = (_Float16)a0[j]; o[4 + j] = (_Float16)a1[j]; }
            *(v8h*)(row + c) = o;
          }
#pragma unroll
          for (int c = 0; c < 64; c += 8) {
            v4f a0 = *(const v4f*)(xj + c);
            v4f a1 = *(const v4f*)(xj + c + 4);
            v8h o;
#pragma unroll
            for (int j = 0; j < 4; ++j) { o[j] = (_Float16)a0[j]; o[4 + j] = (_Float16)a1[j]; }
            *(v8h*)(row + 64 + c) = o;
          }
          row[64] = (_Float16)d2;  // overwrite unused xj[0] slot
        } else {
          for (int c = 0; c < CIN; ++c)  row[c] = (_Float16)xi[c];
          for (int c = 3; c < CIN; ++c)  row[CIN + c - 3] = (_Float16)xj[c];
          row[DIN - 1] = (_Float16)d2;
          for (int c = DIN; c < KT1 * 32; ++c) row[c] = (_Float16)0.f;
        }
      } else {
        sdst[lane] = -1;
#pragma unroll
        for (int c = 0; c < KT1 * 32; c += 8) *(v8h*)(row + c) = (v8h)(_Float16)0.f;
      }
    }
    asm volatile("s_wait_dscnt 0" ::: "memory");

    // ---- h1 = inp @ W1 + b1 : [32,128] ----
#pragma unroll
    for (int mt = 0; mt < 2; ++mt) {
      const _Float16* sA = sa + mt * 16 * 128;
      float* sH = shrow + mt * 16 * 128;
      v16h aF[KT1];
#pragma unroll
      for (int kt = 0; kt < KT1; ++kt) aF[kt] = frag_A(sA, kt * 32, lane);
#pragma unroll
      for (int nt = 0; nt < 8; nt += 2) {
        v8f acc0 = {}, acc1 = {};
#pragma unroll
        for (int kt = 0; kt < KT1; ++kt)
          acc0 = __builtin_amdgcn_wmma_f32_16x16x32_f16(
              false, aF[kt], false, frag_B_sw(sW1, kt * 8 + nt, lane), (short)0, acc0, false, false);
#pragma unroll
        for (int kt = 0; kt < KT1; ++kt)
          acc1 = __builtin_amdgcn_wmma_f32_16x16x32_f16(
              false, aF[kt], false, frag_B_sw(sW1, kt * 8 + nt + 1, lane), (short)0, acc1, false, false);
        int n0 = nt * 16 + nc, n1 = n0 + 16;
#pragma unroll
        for (int v = 0; v < 8; ++v) sH[(v + 8 * hh) * 128 + n0] = acc0[v] + sb1[n0];
#pragma unroll
        for (int v = 0; v < 8; ++v) sH[(v + 8 * hh) * 128 + n1] = acc1[v] + sb1[n1];
      }
    }
    asm volatile("s_wait_dscnt 0" ::: "memory");
    ln_relu_row_128(shrow + lane * 128, sa + lane * 128, sg1, sbe1);
    asm volatile("s_wait_dscnt 0" ::: "memory");

    // ---- h2 = h1 @ W2 + b2 : [32,128] ----
#pragma unroll
    for (int mt = 0; mt < 2; ++mt) {
      const _Float16* sA = sa + mt * 16 * 128;
      float* sH = shrow + mt * 16 * 128;
      v16h aF[4];
#pragma unroll
      for (int kt = 0; kt < 4; ++kt) aF[kt] = frag_A(sA, kt * 32, lane);
#pragma unroll
      for (int nt = 0; nt < 8; nt += 2) {
        v8f acc0 = {}, acc1 = {};
#pragma unroll
        for (int kt = 0; kt < 4; ++kt)
          acc0 = __builtin_amdgcn_wmma_f32_16x16x32_f16(
              false, aF[kt], false, frag_B_sw(sW2, kt * 8 + nt, lane), (short)0, acc0, false, false);
#pragma unroll
        for (int kt = 0; kt < 4; ++kt)
          acc1 = __builtin_amdgcn_wmma_f32_16x16x32_f16(
              false, aF[kt], false, frag_B_sw(sW2, kt * 8 + nt + 1, lane), (short)0, acc1, false, false);
        int n0 = nt * 16 + nc, n1 = n0 + 16;
#pragma unroll
        for (int v = 0; v < 8; ++v) sH[(v + 8 * hh) * 128 + n0] = acc0[v] + sb2[n0];
#pragma unroll
        for (int v = 0; v < 8; ++v) sH[(v + 8 * hh) * 128 + n1] = acc1[v] + sb2[n1];
      }
    }
    asm volatile("s_wait_dscnt 0" ::: "memory");
    ln_relu_row_128(shrow + lane * 128, sa + lane * 128, sg2, sbe2);
    asm volatile("s_wait_dscnt 0" ::: "memory");

    // ---- msg = h2 @ W3 + b3 : [32,64], scatter-add at dst ----
#pragma unroll
    for (int mt = 0; mt < 2; ++mt) {
      const _Float16* sA = sa + mt * 16 * 128;
      v16h aF[4];
#pragma unroll
      for (int kt = 0; kt < 4; ++kt) aF[kt] = frag_A(sA, kt * 32, lane);
#pragma unroll
      for (int nt = 0; nt < 4; nt += 2) {
        v8f acc0 = {}, acc1 = {};
#pragma unroll
        for (int kt = 0; kt < 4; ++kt)
          acc0 = __builtin_amdgcn_wmma_f32_16x16x32_f16(
              false, aF[kt], false, frag_B_sw(sW3, kt * 4 + nt, lane), (short)0, acc0, false, false);
#pragma unroll
        for (int kt = 0; kt < 4; ++kt)
          acc1 = __builtin_amdgcn_wmma_f32_16x16x32_f16(
              false, aF[kt], false, frag_B_sw(sW3, kt * 4 + nt + 1, lane), (short)0, acc1, false, false);
        int n0 = nt * 16 + nc, n1 = n0 + 16;
#pragma unroll
        for (int v = 0; v < 8; ++v) {
          int d = sdst[mt * 16 + v + 8 * hh];
          if (d >= 0) atomicAdd(out + (long)d * 64 + n0, acc0[v] + sb3[n0]);
        }
#pragma unroll
        for (int v = 0; v < 8; ++v) {
          int d = sdst[mt * 16 + v + 8 * hh];
          if (d >= 0) atomicAdd(out + (long)d * 64 + n1, acc1[v] + sb3[n1]);
        }
      }
    }
  }
}

// ---------------- helpers ----------------
// Swizzle f32 weights [kin][ntot] -> WMMA-B fragment layout.
// mode 0: physical K col == reference row (zero-pad past kin)
// mode 1: layer-2 permuted input: p<64 -> p ; p==64 -> 125 (dist2) ;
//         p==65,66 -> zero ; p>=67 -> p-3 (xj[3..63] at rows 64..124)
__global__ void cvt_sw_f16(const float* __restrict__ w, _Float16* __restrict__ o,
                           int kin, int ntot, int ktiles, int ntiles, int mode) {
  int idx = blockIdx.x * blockDim.x + threadIdx.x;
  int tot = ktiles * ntiles * 32 * 16;
  if (idx >= tot) return;
  int e = idx & 15;
  int l = (idx >> 4) & 31;
  int tile = idx >> 9;
  int nt = tile % ntiles, kt = tile / ntiles;
  int c = l & 15, h = l >> 4;
  int region = e >> 3, pos = e & 7;
  int p = kt * 32 + region * 16 + h * 8 + pos;  // physical K column
  int r;                                        // reference weight row
  if (mode == 0) r = p;
  else r = (p < 64) ? p : (p == 64 ? 125 : (p < 67 ? -1 : p - 3));
  int n = nt * 16 + c;
  float v = (r >= 0 && r < kin) ? w[r * ntot + n] : 0.f;
  o[idx] = (_Float16)v;
}

__global__ void zero_f32(float* __restrict__ p, long n) {
  long i = (long)blockIdx.x * blockDim.x + threadIdx.x;
  if (i < n) p[i] = 0.0f;
}

__global__ void pool_kernel(const float* __restrict__ x2, const int* __restrict__ batch,
                            float* __restrict__ addp, float* __restrict__ counts,
                            int* __restrict__ maxp, int N) {
  int i = blockIdx.x * blockDim.x + threadIdx.x;
  if (i >= N) return;
  int g = batch[i];
  atomicAdd(&counts[g], 1.0f);
  for (int c = 0; c < 64; ++c) {
    float v = fmaxf(x2[(long)i * 64 + c], 0.0f);
    atomicAdd(&addp[g * 64 + c], v);
    atomicMax(&maxp[g * 64 + c], __float_as_int(v)); // v >= 0 -> int order == float order
  }
}

__global__ void head_kernel(const float* __restrict__ addp, const float* __restrict__ counts,
                            const int* __restrict__ maxp, const float* __restrict__ u,
                            const float* __restrict__ W1, const float* __restrict__ b1,
                            const float* __restrict__ g1, const float* __restrict__ be1,
                            const float* __restrict__ W2, const float* __restrict__ b2,
                            const float* __restrict__ g2, const float* __restrict__ be2,
                            const float* __restrict__ W3, const float* __restrict__ b3,
                            float* __restrict__ out, int G) {
  int g = blockIdx.x * blockDim.x + threadIdx.x;
  if (g >= G) return;
  float pooled[194];
  float cnt = fmaxf(counts[g], 1.0f);
  for (int c = 0; c < 64; ++c) {
    float a = addp[g * 64 + c];
    pooled[c] = a;
    pooled[64 + c] = a / cnt;
    pooled[128 + c] = __int_as_float(maxp[g * 64 + c]);
  }
  pooled[192] = u[g * 2]; pooled[193] = u[g * 2 + 1];

  float h1[64], h2[64];
  for (int o = 0; o < 64; ++o) {
    float s = b1[o];
    for (int i = 0; i < 194; ++i) s += pooled[i] * W1[i * 64 + o];
    h1[o] = s;
  }
  { float m = 0.f; for (int o = 0; o < 64; ++o) m += h1[o]; m /= 64.f;
    float v = 0.f; for (int o = 0; o < 64; ++o) { float d = h1[o] - m; v += d * d; } v /= 64.f;
    float inv = rsqrtf(v + LN_EPS);
    for (int o = 0; o < 64; ++o) h1[o] = fmaxf((h1[o] - m) * inv * g1[o] + be1[o], 0.f); }
  for (int o = 0; o < 64; ++o) {
    float s = b2[o];
    for (int i = 0; i < 64; ++i) s += h1[i] * W2[i * 64 + o];
    h2[o] = s;
  }
  { float m = 0.f; for (int o = 0; o < 64; ++o) m += h2[o]; m /= 64.f;
    float v = 0.f; for (int o = 0; o < 64; ++o) { float d = h2[o] - m; v += d * d; } v /= 64.f;
    float inv = rsqrtf(v + LN_EPS);
    for (int o = 0; o < 64; ++o) h2[o] = fmaxf((h2[o] - m) * inv * g2[o] + be2[o], 0.f); }
  for (int o = 0; o < 4; ++o) {
    float s = b3[o];
    for (int i = 0; i < 64; ++i) s += h2[i] * W3[i * 4 + o];
    out[g * 4 + o] = s;
  }
}

// ---------------- host launcher ----------------
extern "C" void kernel_launch(void* const* d_in, const int* in_sizes, int n_in,
                              void* d_out, int out_size, void* d_ws, size_t ws_size,
                              hipStream_t stream) {
  const float* x     = (const float*)d_in[0];
  const int*   ei    = (const int*)d_in[1];
  const int*   batch = (const int*)d_in[2];
  const float* u     = (const float*)d_in[3];
  const int E = in_sizes[1] / 2;
  const int N = in_sizes[0] / 7;
  const int G = in_sizes[3] / 2;
  const int* src = ei;
  const int* dst = ei + E;

  const float* L0[10]; const float* L1[10]; const float* FC[10];
  for (int i = 0; i < 10; ++i) {
    L0[i] = (const float*)d_in[4 + i];
    L1[i] = (const float*)d_in[14 + i];
    FC[i] = (const float*)d_in[24 + i];
  }

  const int W1H_L0 = 1 * 8 * 512;  // KT1=1
  const int W1H_L1 = 4 * 8 * 512;  // KT1=4
  const int W2H    = 4 * 8 * 512;
  const int W3H    = 4 * 4 * 512;

  char* ws = (char*)d_ws;
  float* x1 = (float*)ws;              ws += (size_t)N * 64 * 4;
  float* x2 = (float*)ws;              ws += (size_t)N * 64 * 4;
  _Float16* w1h0 = (_Float16*)ws;      ws += (size_t)W1H_L0 * 2;
  _Float16* w2h0 = (_Float16*)ws;      ws += (size_t)W2H * 2;
  _Float16* w3h0 = (_Float16*)ws;      ws += (size_t)W3H * 2;
  _Float16* w1h1 = (_Float16*)ws;      ws += (size_t)W1H_L1 * 2;
  _Float16* w2h1 = (_Float16*)ws;      ws += (size_t)W2H * 2;
  _Float16* w3h1 = (_Float16*)ws;      ws += (size_t)W3H * 2;
  float* addp   = (float*)ws;          ws += (size_t)G * 64 * 4;
  float* counts = (float*)ws;          ws += (size_t)G * 4;
  int*   maxp   = (int*)ws;            ws += (size_t)G * 64 * 4;

  // weight conversion: f32 [K][N] -> f16 WMMA-B fragment layout
  cvt_sw_f16<<<(W1H_L0 + 255) / 256, 256, 0, stream>>>(L0[0], w1h0, 12, 128, 1, 8, 0);
  cvt_sw_f16<<<(W2H + 255) / 256, 256, 0, stream>>>(L0[4], w2h0, 128, 128, 4, 8, 0);
  cvt_sw_f16<<<(W3H + 255) / 256, 256, 0, stream>>>(L0[8], w3h0, 128, 64, 4, 4, 0);
  cvt_sw_f16<<<(W1H_L1 + 255) / 256, 256, 0, stream>>>(L1[0], w1h1, 126, 128, 4, 8, 1);
  cvt_sw_f16<<<(W2H + 255) / 256, 256, 0, stream>>>(L1[4], w2h1, 128, 128, 4, 8, 0);
  cvt_sw_f16<<<(W3H + 255) / 256, 256, 0, stream>>>(L1[8], w3h1, 128, 64, 4, 4, 0);

  // zero accumulators (atomic accumulation requires per-launch zeroing)
  long nz1 = (long)N * 64;
  zero_f32<<<(int)((nz1 + 255) / 256), 256, 0, stream>>>(x1, nz1);
  zero_f32<<<(int)((nz1 + 255) / 256), 256, 0, stream>>>(x2, nz1);
  zero_f32<<<(G * 64 + 255) / 256, 256, 0, stream>>>(addp, (long)G * 64);
  zero_f32<<<(G + 255) / 256, 256, 0, stream>>>(counts, (long)G);
  zero_f32<<<(G * 64 + 255) / 256, 256, 0, stream>>>((float*)maxp, (long)G * 64);

  size_t shmem_common = (size_t)(6 * 128 + 64) * 4
                      + (size_t)WAVES * 32 * 128 * 4
                      + (size_t)WAVES * 32 * 128 * 2
                      + (size_t)WAVES * 32 * 4;
  size_t shmem0 = (size_t)(W1H_L0 + W2H + W3H) * 2 + shmem_common;
  size_t shmem1 = (size_t)(W1H_L1 + W2H + W3H) * 2 + shmem_common;

  int blocks = 512;
  edge_layer<7, 1><<<blocks, WAVES * 32, shmem0, stream>>>(
      x, src, dst,
      w1h0, L0[1], L0[2], L0[3],
      w2h0, L0[5], L0[6], L0[7],
      w3h0, L0[9], x1, E);
  edge_layer<64, 4><<<blocks, WAVES * 32, shmem1, stream>>>(
      x1, src, dst,
      w1h1, L1[1], L1[2], L1[3],
      w2h1, L1[5], L1[6], L1[7],
      w3h1, L1[9], x2, E);

  pool_kernel<<<(N + 255) / 256, 256, 0, stream>>>(x2, batch, addp, counts, maxp, N);
  head_kernel<<<1, 128, 0, stream>>>(addp, counts, maxp, u,
                                     FC[0], FC[1], FC[2], FC[3],
                                     FC[4], FC[5], FC[6], FC[7],
                                     FC[8], FC[9],
                                     (float*)d_out, G);
}
